// Generator_26920855011921
// MI455X (gfx1250) — matrix operand order repaired and checked by reading, not compiled, not run on hardware
//
#include <hip/hip_runtime.h>
#include <hip/hip_bf16.h>

// ---------------------------------------------------------------------------
// CDNA5 (gfx1250) bf16 WMMA implementation of the LSTM generator.
// Matmuls run on v_wmma_f32_16x16x32_bf16 (f32 accumulate); tiles are staged
// with GLOBAL_LOAD_ASYNC_TO_LDS_B128 (double-buffered) when available.
// ---------------------------------------------------------------------------

typedef __attribute__((ext_vector_type(16))) __bf16 v16bf;
typedef __attribute__((ext_vector_type(8)))  __bf16 v8bf;
typedef __attribute__((ext_vector_type(8)))  float  v8f;

// exact pointee type the async builtin expects (gcc-style vector_size)
typedef int v4i __attribute__((vector_size(16)));
typedef __attribute__((address_space(1))) v4i* gv4i_t;   // global (__device__)
typedef __attribute__((address_space(3))) v4i* lv4i_t;   // LDS (__shared__)

#define BM  128
#define BN  64
#define BKL 64   // K tile for fast kernel (2 wmma K-steps per stage)
#define BKG 32   // K tile for guarded kernel

#if defined(__has_builtin)
#if __has_builtin(__builtin_amdgcn_global_load_async_to_lds_b128)
#define HAVE_ASYNC_COPY 1
#endif
#if __has_builtin(__builtin_amdgcn_s_wait_asynccnt)
#define HAVE_WAIT_ASYNC 1
#endif
#endif

__device__ __forceinline__ void async_copy16(const void* g, void* l) {
#ifdef HAVE_ASYNC_COPY
  __builtin_amdgcn_global_load_async_to_lds_b128(
      (gv4i_t)(uintptr_t)g, (lv4i_t)(uintptr_t)l, 0, 0);
#else
  *(v8bf*)l = *(const v8bf*)g;
#endif
}

__device__ __forceinline__ void async_wait0() {
#ifdef HAVE_ASYNC_COPY
#ifdef HAVE_WAIT_ASYNC
  __builtin_amdgcn_s_wait_asynccnt(0);
#else
  asm volatile("s_wait_asynccnt 0x0" ::: "memory");
#endif
#endif
}

// ---- fragment loads (ISA 7.12.2 layouts, wave32) ---------------------------
// A (16x32 bf16): lane l<16 holds row M=l, K in [h*8,h*8+8) u [16+h*8,16+h*8+8)
__device__ __forceinline__ v16bf load_a_frag(const __bf16* As, int ld, int mbase,
                                             int kk, int lane) {
  int r  = lane & 15;
  int hf = lane >> 4;
  const __bf16* p = As + (size_t)(mbase + r) * ld + kk;
  v8bf lo = *(const v8bf*)(p + hf * 8);
  v8bf hi = *(const v8bf*)(p + 16 + hf * 8);
  return __builtin_shufflevector(lo, hi, 0,1,2,3,4,5,6,7,8,9,10,11,12,13,14,15);
}

// B (32x16 bf16): lane holds column N=lane&15, K in [h*8,h*8+8) u [16+h*8,16+h*8+8)
__device__ __forceinline__ v16bf load_b_frag(const __bf16* Bs, int ld, int nbase,
                                             int kk, int lane) {
  int n  = lane & 15;
  int kh = lane >> 4;
  const __bf16* p = Bs + (size_t)(nbase + n) * ld + kk;
  v8bf lo = *(const v8bf*)(p + kh * 8);
  v8bf hi = *(const v8bf*)(p + 16 + kh * 8);
  return __builtin_shufflevector(lo, hi, 0,1,2,3,4,5,6,7,8,9,10,11,12,13,14,15);
}

// ---- fast (unguarded) dual GEMM: C = A0·B0^T [+ A1·B1^T] (+bias)(+act) -----
// Requires: M % BM == 0, N % BN == 0, K % BKL == 0.
// mode 0: store fp32 to Cf.   mode 1: LeakyReLU(0.01) then store bf16 to Cb.
__global__ __launch_bounds__(256) void gemm_wmma_fast(
    const __bf16* __restrict__ A0, const __bf16* __restrict__ B0, int K0,
    const __bf16* __restrict__ A1, const __bf16* __restrict__ B1, int K1,
    float* __restrict__ Cf, __bf16* __restrict__ Cb,
    const float* __restrict__ bias, int M, int N, int mode)
{
  __shared__ __bf16 As[2][BM * BKL];   // 2 x 16 KB
  __shared__ __bf16 Bs[2][BN * BKL];   // 2 x  8 KB

  const int tid  = threadIdx.x;
  const int lane = tid & 31;
  const int wave = tid >> 5;
  const int wm   = wave >> 1;   // 0..3  (M direction, 32 rows each)
  const int wn   = wave & 1;    // 0..1  (N direction, 32 cols each)
  const int m0   = blockIdx.y * BM;
  const int n0   = blockIdx.x * BN;

  v8f acc[2][2];
#pragma unroll
  for (int i = 0; i < 2; ++i)
#pragma unroll
    for (int j = 0; j < 2; ++j)
#pragma unroll
      for (int q = 0; q < 8; ++q) acc[i][j][q] = 0.0f;

  const int s0  = K0 / BKL;
  const int s1  = A1 ? (K1 / BKL) : 0;
  const int nst = s0 + s1;

  auto stage = [&](int s, int buf) {
    const __bf16* A = (s < s0) ? A0 : A1;
    const __bf16* B = (s < s0) ? B0 : B1;
    const int     K = (s < s0) ? K0 : K1;
    const int    k0 = ((s < s0) ? s : (s - s0)) * BKL;
    // A tile: 128 x 64 bf16 = 1024 x 16B chunks; 4 per thread
#pragma unroll
    for (int p = 0; p < 4; ++p) {
      int idx = tid + p * 256;
      int r   = idx >> 3;
      int cc  = (idx & 7) * 8;
      async_copy16(A + (size_t)(m0 + r) * K + k0 + cc, &As[buf][r * BKL + cc]);
    }
    // B tile: 64 x 64 bf16 = 512 x 16B chunks; 2 per thread
#pragma unroll
    for (int p = 0; p < 2; ++p) {
      int idx = tid + p * 256;
      int r   = idx >> 3;
      int cc  = (idx & 7) * 8;
      async_copy16(B + (size_t)(n0 + r) * K + k0 + cc, &Bs[buf][r * BKL + cc]);
    }
  };

  stage(0, 0);
  async_wait0();
  __syncthreads();

  for (int s = 0; s < nst; ++s) {
    const int cur = s & 1;
    if (s + 1 < nst) stage(s + 1, cur ^ 1);   // overlap fetch with compute

#pragma unroll
    for (int kk = 0; kk < BKL; kk += 32) {
      v16bf af[2], bfr[2];
#pragma unroll
      for (int i = 0; i < 2; ++i)
        af[i]  = load_a_frag(&As[cur][0], BKL, wm * 32 + i * 16, kk, lane);
#pragma unroll
      for (int j = 0; j < 2; ++j)
        bfr[j] = load_b_frag(&Bs[cur][0], BKL, wn * 32 + j * 16, kk, lane);
#pragma unroll
      for (int i = 0; i < 2; ++i)
#pragma unroll
        for (int j = 0; j < 2; ++j)
          acc[i][j] = __builtin_amdgcn_wmma_f32_16x16x32_bf16(
              false, af[i], false, bfr[j], (short)0, acc[i][j], false, false);
    }
    async_wait0();     // own async copies into other buffer complete
    __syncthreads();   // everyone done reading cur + copies visible
  }

  // epilogue: C/D layout -> M = v + 8*(lane>=16), N = lane&15
  const int nn = lane & 15;
  const int hf = lane >> 4;
#pragma unroll
  for (int i = 0; i < 2; ++i)
#pragma unroll
    for (int j = 0; j < 2; ++j)
#pragma unroll
      for (int v = 0; v < 8; ++v) {
        int gm = m0 + wm * 32 + i * 16 + v + hf * 8;
        int gn = n0 + wn * 32 + j * 16 + nn;
        float x = acc[i][j][v];
        if (bias) x += bias[gn];
        if (mode == 1) {
          x = x > 0.0f ? x : 0.01f * x;
          Cb[(size_t)gm * N + gn] = (__bf16)x;
        } else {
          Cf[(size_t)gm * N + gn] = x;
        }
      }
}

// ---- guarded GEMM (for ragged N, e.g. final N=100): C = A·B^T + bias -------
__global__ __launch_bounds__(256) void gemm_wmma_guarded(
    const __bf16* __restrict__ A, const __bf16* __restrict__ B, int K,
    float* __restrict__ Cf, const float* __restrict__ bias, int M, int N)
{
  __shared__ __bf16 As[BM * BKG];
  __shared__ __bf16 Bs[BN * BKG];

  const int tid  = threadIdx.x;
  const int lane = tid & 31;
  const int wave = tid >> 5;
  const int wm   = wave >> 1;
  const int wn   = wave & 1;
  const int m0   = blockIdx.y * BM;
  const int n0   = blockIdx.x * BN;

  v8f acc[2][2];
#pragma unroll
  for (int i = 0; i < 2; ++i)
#pragma unroll
    for (int j = 0; j < 2; ++j)
#pragma unroll
      for (int q = 0; q < 8; ++q) acc[i][j][q] = 0.0f;

  for (int k0 = 0; k0 < K; k0 += BKG) {
#pragma unroll
    for (int p = 0; p < 2; ++p) {
      int r  = (tid >> 2) + p * 64;
      int cc = (tid & 3) * 8;
      int gm = m0 + r;
      v8bf v;
#pragma unroll
      for (int q = 0; q < 8; ++q) v[q] = (__bf16)0.0f;
      if (gm < M) v = *(const v8bf*)(A + (size_t)gm * K + k0 + cc);
      *(v8bf*)(As + (size_t)r * BKG + cc) = v;
    }
    {
      int r  = tid >> 2;
      int cc = (tid & 3) * 8;
      int gn = n0 + r;
      v8bf v;
#pragma unroll
      for (int q = 0; q < 8; ++q) v[q] = (__bf16)0.0f;
      if (gn < N) v = *(const v8bf*)(B + (size_t)gn * K + k0 + cc);
      *(v8bf*)(Bs + (size_t)r * BKG + cc) = v;
    }
    __syncthreads();

    v16bf af[2], bfr[2];
#pragma unroll
    for (int i = 0; i < 2; ++i) af[i]  = load_a_frag(As, BKG, wm * 32 + i * 16, 0, lane);
#pragma unroll
    for (int j = 0; j < 2; ++j) bfr[j] = load_b_frag(Bs, BKG, wn * 32 + j * 16, 0, lane);
#pragma unroll
    for (int i = 0; i < 2; ++i)
#pragma unroll
      for (int j = 0; j < 2; ++j)
        acc[i][j] = __builtin_amdgcn_wmma_f32_16x16x32_bf16(
            false, af[i], false, bfr[j], (short)0, acc[i][j], false, false);
    __syncthreads();
  }

  const int nn = lane & 15;
  const int hf = lane >> 4;
#pragma unroll
  for (int i = 0; i < 2; ++i)
#pragma unroll
    for (int j = 0; j < 2; ++j)
#pragma unroll
      for (int v = 0; v < 8; ++v) {
        int gm = m0 + wm * 32 + i * 16 + v + hf * 8;
        int gn = n0 + wn * 32 + j * 16 + nn;
        if (gm < M && gn < N) Cf[(size_t)gm * N + gn] = acc[i][j][v] + bias[gn];
      }
}

// ---- fp32 -> bf16 conversion ----------------------------------------------
__global__ void cvt_bf16(const float* __restrict__ src, __bf16* __restrict__ dst, int n) {
  int i = blockIdx.x * blockDim.x + threadIdx.x;
  if (i < n) dst[i] = (__bf16)src[i];
}

// ---- BatchNorm over small batch (R=256): one thread per column -------------
__global__ void bn_small(const float* __restrict__ in, const float* __restrict__ g,
                         const float* __restrict__ b, __bf16* __restrict__ out_bf,
                         float* __restrict__ out_f, int R, int C)
{
  int col = blockIdx.x * blockDim.x + threadIdx.x;
  if (col >= C) return;
  float s = 0.f, s2 = 0.f;
  for (int r = 0; r < R; ++r) {
    float v = in[(size_t)r * C + col];
    s += v; s2 += v * v;
  }
  float mean = s / (float)R;
  float var  = s2 / (float)R - mean * mean;
  float rstd = rsqrtf(var + 1e-5f);
  float a  = g[col] * rstd;
  float bb = b[col] - mean * a;
  for (int r = 0; r < R; ++r) {
    float v = in[(size_t)r * C + col] * a + bb;
    if (out_bf) out_bf[(size_t)r * C + col] = (__bf16)v;
    else        out_f [(size_t)r * C + col] = v;
  }
}

// ---- BatchNorm over R=65536 rows, in place on bf16, 64 columns per block ---
__global__ __launch_bounds__(256) void bn_big(__bf16* __restrict__ data,
                                              const float* __restrict__ g,
                                              const float* __restrict__ b,
                                              int R, int C)
{
  __shared__ float s1[4][64], s2[4][64], sa[64], sb[64];
  const int cidx = threadIdx.x & 63;
  const int rg   = threadIdx.x >> 6;
  const int c    = blockIdx.x * 64 + cidx;
  float p1 = 0.f, p2 = 0.f;
  for (int r = rg; r < R; r += 4) {
    float v = (float)data[(size_t)r * C + c];
    p1 += v; p2 += v * v;
  }
  s1[rg][cidx] = p1; s2[rg][cidx] = p2;
  __syncthreads();
  if (rg == 0) {
    float t1 = s1[0][cidx] + s1[1][cidx] + s1[2][cidx] + s1[3][cidx];
    float t2 = s2[0][cidx] + s2[1][cidx] + s2[2][cidx] + s2[3][cidx];
    float mean = t1 / (float)R;
    float var  = t2 / (float)R - mean * mean;
    float rstd = rsqrtf(var + 1e-5f);
    float a = g[c] * rstd;
    sa[cidx] = a; sb[cidx] = b[c] - mean * a;
  }
  __syncthreads();
  float a = sa[cidx], bb = sb[cidx];
  for (int r = rg; r < R; r += 4) {
    float v = (float)data[(size_t)r * C + c];
    data[(size_t)r * C + c] = (__bf16)(v * a + bb);
  }
}

// ---- LSTM cell pointwise: gate order i,f,g,o over N=4096 -------------------
__global__ __launch_bounds__(256) void lstm_cell(
    const float* __restrict__ gates, const float* __restrict__ b_ih,
    const float* __restrict__ b_hh, float* __restrict__ c,
    __bf16* __restrict__ h_bf, __bf16* __restrict__ x_bf,
    __bf16* __restrict__ ys, int t)
{
  int idx = blockIdx.x * blockDim.x + threadIdx.x;       // 0 .. 256*1024
  int bb  = idx >> 10;
  int j   = idx & 1023;
  size_t base = (size_t)bb * 4096;
  float gi = gates[base +        j] + b_ih[       j] + b_hh[       j];
  float gf = gates[base + 1024 + j] + b_ih[1024 + j] + b_hh[1024 + j];
  float gg = gates[base + 2048 + j] + b_ih[2048 + j] + b_hh[2048 + j];
  float go = gates[base + 3072 + j] + b_ih[3072 + j] + b_hh[3072 + j];
  float i_ = 1.0f / (1.0f + __expf(-gi));
  float f_ = 1.0f / (1.0f + __expf(-gf));
  float g_ = tanhf(gg);
  float o_ = 1.0f / (1.0f + __expf(-go));
  float c2 = f_ * c[idx] + i_ * g_;
  float h2 = o_ * tanhf(c2);
  c[idx] = c2;
  __bf16 hb = (__bf16)h2;
  h_bf[idx] = hb;
  x_bf[idx] = hb;
  ys[(size_t)bb * 1024 * 256 + (size_t)j * 256 + t] = hb;   // [B, H, T]
}

// ---- softmax over axis 1 of [256, 100, 256], in place ----------------------
__global__ void softmax_axis1(float* __restrict__ out) {
  int idx = blockIdx.x * blockDim.x + threadIdx.x;   // 65536 (b,t) pairs
  int b = idx >> 8;
  int t = idx & 255;
  float* p = out + (size_t)b * 100 * 256 + t;
  float mx = -3.0e38f;
  for (int n = 0; n < 100; ++n) mx = fmaxf(mx, p[(size_t)n * 256]);
  float s = 0.f;
  for (int n = 0; n < 100; ++n) {
    float e = __expf(p[(size_t)n * 256] - mx);
    p[(size_t)n * 256] = e;
    s += e;
  }
  float inv = 1.0f / s;
  for (int n = 0; n < 100; ++n) p[(size_t)n * 256] *= inv;
}

// ---------------------------------------------------------------------------
extern "C" void kernel_launch(void* const* d_in, const int* in_sizes, int n_in,
                              void* d_out, int out_size, void* d_ws, size_t ws_size,
                              hipStream_t stream) {
  (void)in_sizes; (void)n_in; (void)out_size; (void)ws_size;
  const float* z      = (const float*)d_in[0];
  const float* W_zh   = (const float*)d_in[1];
  const float* b_zh   = (const float*)d_in[2];
  const float* W_zc   = (const float*)d_in[3];
  const float* b_zc   = (const float*)d_in[4];
  const float* g_bnzh = (const float*)d_in[5];
  const float* b_bnzh = (const float*)d_in[6];
  const float* g_bnzc = (const float*)d_in[7];
  const float* b_bnzc = (const float*)d_in[8];
  const float* W_ih   = (const float*)d_in[9];
  const float* b_ih   = (const float*)d_in[10];
  const float* W_hh   = (const float*)d_in[11];
  const float* b_hh   = (const float*)d_in[12];
  const float* g_bn0  = (const float*)d_in[13];
  const float* b_bn0  = (const float*)d_in[14];
  const float* W1     = (const float*)d_in[15];
  const float* b1     = (const float*)d_in[16];
  const float* g_bn1  = (const float*)d_in[17];
  const float* b_bn1  = (const float*)d_in[18];
  const float* W2     = (const float*)d_in[19];
  const float* b2     = (const float*)d_in[20];
  const float* g_bn2  = (const float*)d_in[21];
  const float* b_bn2  = (const float*)d_in[22];
  const float* W3     = (const float*)d_in[23];
  const float* b3     = (const float*)d_in[24];
  const float* go     = (const float*)d_in[25];

  const int Bsz = 256, T = 256, H = 1024, NZ = 128, NOUT = 100;
  const int RB = Bsz * T;   // 65536 head rows

  char*  base = (char*)d_ws;
  size_t off  = 0;
  auto alloc = [&](size_t bytes) -> char* {
    char* p = base + off;
    off = (off + bytes + 255) & ~(size_t)255;
    return p;
  };
  __bf16* Wih_bf = (__bf16*)alloc((size_t)4 * H * H * 2);
  __bf16* Whh_bf = (__bf16*)alloc((size_t)4 * H * H * 2);
  __bf16* Wzh_bf = (__bf16*)alloc((size_t)H * NZ * 2);
  __bf16* Wzc_bf = (__bf16*)alloc((size_t)H * NZ * 2);
  __bf16* W1_bf  = (__bf16*)alloc((size_t)512 * H * 2);
  __bf16* W2_bf  = (__bf16*)alloc((size_t)256 * 512 * 2);
  __bf16* W3_bf  = (__bf16*)alloc((size_t)NOUT * 256 * 2);
  __bf16* z_bf   = (__bf16*)alloc((size_t)Bsz * NZ * 2);
  __bf16* x_bf   = (__bf16*)alloc((size_t)Bsz * H * 2);
  __bf16* h_bf   = (__bf16*)alloc((size_t)Bsz * H * 2);
  float*  c_st   = (float*) alloc((size_t)Bsz * H * 4);
  float*  zh_f   = (float*) alloc((size_t)Bsz * H * 4);
  float*  zc_f   = (float*) alloc((size_t)Bsz * H * 4);
  float*  gates  = (float*) alloc((size_t)Bsz * 4 * H * 4);
  __bf16* ys_bf  = (__bf16*)alloc((size_t)Bsz * H * T * 2);
  __bf16* o1_bf  = (__bf16*)alloc((size_t)RB * 512 * 2);
  __bf16* o2_bf  = (__bf16*)alloc((size_t)RB * 256 * 2);

  auto CVT = [&](const float* s, __bf16* d, int n) {
    cvt_bf16<<<(n + 255) / 256, 256, 0, stream>>>(s, d, n);
  };
  CVT(W_ih, Wih_bf, 4 * H * H);
  CVT(W_hh, Whh_bf, 4 * H * H);
  CVT(W_zh, Wzh_bf, H * NZ);
  CVT(W_zc, Wzc_bf, H * NZ);
  CVT(W1,   W1_bf,  512 * H);
  CVT(W2,   W2_bf,  256 * 512);
  CVT(W3,   W3_bf,  NOUT * 256);
  CVT(z,    z_bf,   Bsz * NZ);
  CVT(go,   x_bf,   Bsz * H);   // initial x = go token

  auto GEMM = [&](const __bf16* A0, const __bf16* B0, int K0,
                  const __bf16* A1, const __bf16* B1, int K1,
                  float* Cf, __bf16* Cb, const float* bias,
                  int M, int N, int mode) {
    dim3 g(N / BN, M / BM);
    gemm_wmma_fast<<<g, 256, 0, stream>>>(A0, B0, K0, A1, B1, K1, Cf, Cb, bias, M, N, mode);
  };

  // initial projections + BN -> h, c
  GEMM(z_bf, Wzh_bf, NZ, nullptr, nullptr, 0, zh_f, nullptr, b_zh, Bsz, H, 0);
  GEMM(z_bf, Wzc_bf, NZ, nullptr, nullptr, 0, zc_f, nullptr, b_zc, Bsz, H, 0);
  bn_small<<<H / 256, 256, 0, stream>>>(zh_f, g_bnzh, b_bnzh, h_bf, nullptr, Bsz, H);
  bn_small<<<H / 256, 256, 0, stream>>>(zc_f, g_bnzc, b_bnzc, nullptr, c_st, Bsz, H);

  // LSTM recurrence: gates = x·W_ih^T + h·W_hh^T  (dual-GEMM), then cell
  for (int t = 0; t < T; ++t) {
    GEMM(x_bf, Wih_bf, H, h_bf, Whh_bf, H, gates, nullptr, nullptr, Bsz, 4 * H, 0);
    lstm_cell<<<(Bsz * H) / 256, 256, 0, stream>>>(gates, b_ih, b_hh, c_st, h_bf, x_bf, ys_bf, t);
  }

  // head: BN0 -> FC1+LeakyReLU -> BN1 -> FC2+LeakyReLU -> BN2 -> FC3 -> softmax
  bn_big<<<H / 64, 256, 0, stream>>>(ys_bf, g_bn0, b_bn0, RB, H);
  GEMM(ys_bf, W1_bf, H, nullptr, nullptr, 0, nullptr, o1_bf, b1, RB, 512, 1);
  bn_big<<<512 / 64, 256, 0, stream>>>(o1_bf, g_bn1, b_bn1, RB, 512);
  GEMM(o1_bf, W2_bf, 512, nullptr, nullptr, 0, nullptr, o2_bf, b2, RB, 256, 1);
  bn_big<<<256 / 64, 256, 0, stream>>>(o2_bf, g_bn2, b_bn2, RB, 256);
  {
    dim3 g((NOUT + BN - 1) / BN, RB / BM);
    gemm_wmma_guarded<<<g, 256, 0, stream>>>(o2_bf, W3_bf, 256, (float*)d_out, b3, RB, NOUT);
  }
  softmax_axis1<<<(Bsz * T) / 256, 256, 0, stream>>>((float*)d_out);
}
